// Attention_63256278335572
// MI455X (gfx1250) — matrix operand order repaired and checked by reading, not compiled, not run on hardware
//
#include <hip/hip_runtime.h>

#define HEAD_NUM   16
#define HEAD_DIM   64
#define INPUT_DIM  1024
#define OUTPUT_DIM 1024
#define SEQ_N      2048
#define BATCH      2
#define QKV_LD     (3 * HEAD_NUM * HEAD_DIM)   // 3072

typedef __attribute__((ext_vector_type(16))) __bf16 v16bf;
typedef __attribute__((ext_vector_type(8)))  float  v8f;

union Frag {
    v16bf v;
    unsigned int u[8];
};

__device__ __forceinline__ unsigned short f2bf(float f) {
    unsigned int u = __float_as_uint(f);
    unsigned int r = u + 0x7FFFu + ((u >> 16) & 1u);   // round-to-nearest-even
    return (unsigned short)(r >> 16);
}

// A-matrix 16x32 bf16 fragment (M in lane%16, K split: lanes<16 -> K 0-15; lanes>=16 -> K 16-31)
__device__ __forceinline__ void load_a_frag(const unsigned short* __restrict__ s,
                                            int rowStride, int m, int kbase, Frag& f) {
    const int koff0 = kbase;
#pragma unroll
    for (int i = 0; i < 4; ++i)
        f.u[i] = *(const unsigned int*)(s + m * rowStride + koff0 + 2 * i);
#pragma unroll
    for (int i = 0; i < 4; ++i)
        f.u[4 + i] = *(const unsigned int*)(s + m * rowStride + koff0 + 8 + 2 * i);
}

// B-matrix 32x16 bf16 fragment from LDS stored as [n][k] (K-pairs contiguous).
// N in lane%16, K split: lanes<16 -> K 0-15; lanes>=16 -> K 16-31.
__device__ __forceinline__ void load_b_frag(const unsigned short* __restrict__ s,
                                            int rowStride, int n, int kbase, Frag& f) {
#pragma unroll
    for (int i = 0; i < 8; ++i)
        f.u[i] = *(const unsigned int*)(s + n * rowStride + kbase + 2 * i);
}

// ---------------------------------------------------------------------------
// Generic GEMM: C[M,Ncols] = A[M,K] * B[K,Ncols] + bias[Ncols]   (f32 in/out,
// bf16 WMMA compute). Block tile 128x128, K-step 32, 8 waves of 32x64 tiles.
// All dims here are multiples of the tile sizes (4096/3072/1024).
// ---------------------------------------------------------------------------
__global__ __launch_bounds__(256) void gemm_bias_wmma(
    const float* __restrict__ A, const float* __restrict__ Bw,
    const float* __restrict__ bias, float* __restrict__ C,
    int K, int lda, int ldb, int ldc) {

    __shared__ __align__(16) unsigned short sA[128 * 34];   // [m][k] pad 2
    __shared__ __align__(16) unsigned short sB[128 * 34];   // [n][k] pad 2

    const int tid  = threadIdx.x;
    const int wave = tid >> 5;
    const int lane = tid & 31;
    const int ln16 = lane & 15;
    const int kbA  = (lane < 16) ? 0 : 16;   // A-frag K base
    const int kbB  = (lane < 16) ? 0 : 16;   // B-frag K base

    const int wm = wave & 3;                 // 4 waves along M  -> 4*32 = 128
    const int wn = wave >> 2;                // 2 waves along N  -> 2*64 = 128
    const int m0 = blockIdx.y * 128;
    const int n0 = blockIdx.x * 128;

    v8f acc[2][4];
#pragma unroll
    for (int a = 0; a < 2; ++a)
#pragma unroll
        for (int b = 0; b < 4; ++b)
            acc[a][b] = (v8f){0.f, 0.f, 0.f, 0.f, 0.f, 0.f, 0.f, 0.f};

    for (int k0 = 0; k0 < K; k0 += 32) {
        __syncthreads();
        // A tile: 128x32, row-major bf16
#pragma unroll
        for (int i = 0; i < 16; ++i) {
            int idx = tid + i * 256;
            int r = idx >> 5, k = idx & 31;
            sA[r * 34 + k] = f2bf(A[(size_t)(m0 + r) * lda + k0 + k]);
        }
        // B tile: 32x128 in memory, stored transposed [n][k]
#pragma unroll
        for (int i = 0; i < 16; ++i) {
            int idx = tid + i * 256;
            int k = idx >> 7, n = idx & 127;
            sB[n * 34 + k] = f2bf(Bw[(size_t)(k0 + k) * ldb + n0 + n]);
        }
        __syncthreads();

        Frag af[2], bf[4];
#pragma unroll
        for (int a = 0; a < 2; ++a)
            load_a_frag(sA, 34, wm * 32 + a * 16 + ln16, kbA, af[a]);
#pragma unroll
        for (int b = 0; b < 4; ++b)
            load_b_frag(sB, 34, wn * 64 + b * 16 + ln16, kbB, bf[b]);

#pragma unroll
        for (int a = 0; a < 2; ++a)
#pragma unroll
            for (int b = 0; b < 4; ++b)
                acc[a][b] = __builtin_amdgcn_wmma_f32_16x16x32_bf16(
                    false, af[a].v, false, bf[b].v, (short)0, acc[a][b], false, false);
    }

    // epilogue: add bias, store f32 (C layout: col = lane%16, row = base + r)
    const int rbase = (lane < 16) ? 0 : 8;
#pragma unroll
    for (int b = 0; b < 4; ++b) {
        int col = n0 + wn * 64 + b * 16 + ln16;
        float bv = bias[col];
#pragma unroll
        for (int a = 0; a < 2; ++a) {
#pragma unroll
            for (int r = 0; r < 8; ++r) {
                int row = m0 + wm * 32 + a * 16 + rbase + r;
                C[(size_t)row * ldc + col] = acc[a][b][r] + bv;
            }
        }
    }
}

// ---------------------------------------------------------------------------
// Attention: one block per (b, h, 64-query tile). 4 waves x 16 query rows.
// Online (flash) softmax over 32-key steps, valid-length masking (-1e9).
// qkv layout: [B][N][3072] with Q at +0, K at +1024, V at +2048, head h at h*64.
// ---------------------------------------------------------------------------
__global__ __launch_bounds__(128) void attention_wmma(
    const float* __restrict__ qkv, const int* __restrict__ vaild,
    float* __restrict__ attnOut) {

    __shared__ __align__(16) unsigned short sQ[64 * 66];        // [qrow][d]
    __shared__ __align__(16) unsigned short sK[32 * 66];        // [key][d]
    __shared__ __align__(16) unsigned short sV[64 * 34];        // [d][key] (transposed)
    __shared__ __align__(16) unsigned short sP[4][16 * 34];     // per-wave [row][keyLocal]

    const int tid  = threadIdx.x;
    const int wave = tid >> 5;
    const int lane = tid & 31;
    const int ln16 = lane & 15;
    const int kbA  = (lane < 16) ? 0 : 16;
    const int kbB  = (lane < 16) ? 0 : 16;
    const int rbase = (lane < 16) ? 0 : 8;

    const int bid = blockIdx.x;
    const int qt = bid & 31;            // 32 query tiles of 64
    const int h  = (bid >> 5) & 15;
    const int b  = bid >> 9;
    const int vn = vaild[b];

    // --- load Q tile (64x64), scale by 1/sqrt(64), bf16 ---
#pragma unroll
    for (int i = 0; i < 32; ++i) {
        int idx = tid + i * 128;
        int r = idx >> 6, d = idx & 63;
        float q = qkv[(size_t)(b * SEQ_N + qt * 64 + r) * QKV_LD + h * HEAD_DIM + d];
        sQ[r * 66 + d] = f2bf(q * 0.125f);
    }
    __syncthreads();

    Frag qf[2];
#pragma unroll
    for (int c = 0; c < 2; ++c)
        load_a_frag(sQ, 66, wave * 16 + ln16, c * 32 + kbA, qf[c]);

    v8f o[4];
#pragma unroll
    for (int f = 0; f < 4; ++f)
        o[f] = (v8f){0.f, 0.f, 0.f, 0.f, 0.f, 0.f, 0.f, 0.f};
    float mrow[8], lrow[8];
#pragma unroll
    for (int r = 0; r < 8; ++r) { mrow[r] = -3.0e38f; lrow[r] = 0.f; }

    bool rowValid[8];
#pragma unroll
    for (int r = 0; r < 8; ++r)
        rowValid[r] = (qt * 64 + wave * 16 + rbase + r) < vn;

    for (int kt = 0; kt < SEQ_N; kt += 32) {
        __syncthreads();
        // K tile 32x64 row-major; V tile 32x64 stored transposed [d][key]
#pragma unroll
        for (int i = 0; i < 16; ++i) {
            int idx = tid + i * 128;
            int key = idx >> 6, d = idx & 63;
            size_t base = (size_t)(b * SEQ_N + kt + key) * QKV_LD + h * HEAD_DIM + d;
            sK[key * 66 + d] = f2bf(qkv[base + 1024]);
            sV[d * 34 + key] = f2bf(qkv[base + 2048]);
        }
        __syncthreads();

        // --- S = Q * K^T for two 16-key sub-tiles ---
        v8f s[2];
#pragma unroll
        for (int t = 0; t < 2; ++t) {
            s[t] = (v8f){0.f, 0.f, 0.f, 0.f, 0.f, 0.f, 0.f, 0.f};
            Frag kf0, kf1;
            load_b_frag(sK, 66, t * 16 + ln16, kbB, kf0);         // d-chunk 0
            load_b_frag(sK, 66, t * 16 + ln16, 32 + kbB, kf1);    // d-chunk 1
            s[t] = __builtin_amdgcn_wmma_f32_16x16x32_bf16(
                false, qf[0].v, false, kf0.v, (short)0, s[t], false, false);
            s[t] = __builtin_amdgcn_wmma_f32_16x16x32_bf16(
                false, qf[1].v, false, kf1.v, (short)0, s[t], false, false);
            // valid-length mask (scale already folded into Q)
            bool colValid = (kt + t * 16 + ln16) < vn;
#pragma unroll
            for (int r = 0; r < 8; ++r)
                if (!(colValid && rowValid[r])) s[t][r] = -1.0e9f;
        }

        // --- online softmax (rows live in VGPR index + lane half; cols in lanes) ---
        float fac[8];
#pragma unroll
        for (int r = 0; r < 8; ++r) {
            float v = fmaxf(s[0][r], s[1][r]);
#pragma unroll
            for (int msk = 8; msk >= 1; msk >>= 1)
                v = fmaxf(v, __shfl_xor(v, msk, 32));
            float mnew = fmaxf(mrow[r], v);
            fac[r] = __expf(mrow[r] - mnew);
            mrow[r] = mnew;
            s[0][r] = __expf(s[0][r] - mnew);
            s[1][r] = __expf(s[1][r] - mnew);
            float rs = s[0][r] + s[1][r];
#pragma unroll
            for (int msk = 8; msk >= 1; msk >>= 1)
                rs += __shfl_xor(rs, msk, 32);
            lrow[r] = lrow[r] * fac[r] + rs;
#pragma unroll
            for (int f = 0; f < 4; ++f)
                o[f][r] *= fac[r];
        }

        // --- re-layout P (C-frag -> A-frag) through per-wave LDS scratch ---
        unsigned short* sPw = sP[wave];
#pragma unroll
        for (int t = 0; t < 2; ++t)
#pragma unroll
            for (int r = 0; r < 8; ++r)
                sPw[(rbase + r) * 34 + t * 16 + ln16] = f2bf(s[t][r]);
        __syncthreads();

        Frag pf;
        load_a_frag(sPw, 34, ln16, kbA, pf);

        // --- O += P * V ---
#pragma unroll
        for (int f = 0; f < 4; ++f) {
            Frag vf;
            load_b_frag(sV, 34, f * 16 + ln16, kbB, vf);
            o[f] = __builtin_amdgcn_wmma_f32_16x16x32_bf16(
                false, pf.v, false, vf.v, (short)0, o[f], false, false);
        }
    }

    // --- epilogue: normalize and store attn [B][N][H*64] f32 ---
#pragma unroll
    for (int r = 0; r < 8; ++r) {
        float inv = 1.0f / lrow[r];
        int qrow = qt * 64 + wave * 16 + rbase + r;
#pragma unroll
        for (int f = 0; f < 4; ++f)
            attnOut[(size_t)(b * SEQ_N + qrow) * (HEAD_NUM * HEAD_DIM)
                    + h * HEAD_DIM + f * 16 + ln16] = o[f][r] * inv;
    }
}

extern "C" void kernel_launch(void* const* d_in, const int* in_sizes, int n_in,
                              void* d_out, int out_size, void* d_ws, size_t ws_size,
                              hipStream_t stream) {
    (void)in_sizes; (void)n_in; (void)out_size; (void)ws_size;
    const float* x      = (const float*)d_in[0];
    const int*   vaild  = (const int*)d_in[1];
    const float* W_qkv  = (const float*)d_in[2];
    const float* b_qkv  = (const float*)d_in[3];
    const float* W_proj = (const float*)d_in[4];
    const float* b_proj = (const float*)d_in[5];
    float* out = (float*)d_out;

    const int M = BATCH * SEQ_N;                       // 4096
    float* qkv  = (float*)d_ws;                        // [4096][3072]
    float* attn = qkv + (size_t)M * QKV_LD;            // [4096][1024]

    // 1) QKV projection: [4096,1024] x [1024,3072] + bias
    {
        dim3 grid(QKV_LD / 128, M / 128);
        gemm_bias_wmma<<<grid, 256, 0, stream>>>(x, W_qkv, b_qkv, qkv,
                                                 INPUT_DIM, INPUT_DIM, QKV_LD, QKV_LD);
    }
    // 2) Attention with masking + softmax
    {
        dim3 grid(BATCH * HEAD_NUM * (SEQ_N / 64));
        attention_wmma<<<grid, 128, 0, stream>>>(qkv, vaild, attn);
    }
    // 3) Output projection: [4096,1024] x [1024,1024] + bias
    {
        dim3 grid(OUTPUT_DIM / 128, M / 128);
        gemm_bias_wmma<<<grid, 256, 0, stream>>>(attn, W_proj, b_proj, out,
                                                 HEAD_NUM * HEAD_DIM,
                                                 OUTPUT_DIM, OUTPUT_DIM, OUTPUT_DIM);
    }
}
